// VectorQuantizer_32719060861528
// MI455X (gfx1250) — matrix operand order repaired and checked by reading, not compiled, not run on hardware
//
#include <hip/hip_runtime.h>
#include <hip/hip_bf16.h>

// Problem constants (from reference): B=64, C=64, H=32, W=32, K=1024
#define VQ_B   64
#define VQ_C   64
#define VQ_HW  1024              // H*W
#define VQ_N   65536             // B*H*W rows
#define VQ_K   1024              // codes
#define OUT_Q  4194304           // B*C*H*W
#define OUT_I  65536             // B*H*W

typedef __attribute__((ext_vector_type(16))) _Float16 v16h;
typedef __attribute__((ext_vector_type(8)))  _Float16 v8h;
typedef __attribute__((ext_vector_type(8)))  float    v8f;

union V16 { v16h v; v8h h[2]; };

// ---------------------------------------------------------------------------
// Kernel 1: weight -> f16 copy + per-code squared norms (f32).
// grid = K blocks of 32 threads (1 wave each).
__global__ void vq_prep(const float* __restrict__ wgt,
                        _Float16* __restrict__ wq,
                        float* __restrict__ norms) {
    int k = blockIdx.x;
    int c = threadIdx.x;           // 0..31
    float w0 = wgt[k * 64 + c];
    float w1 = wgt[k * 64 + 32 + c];
    wq[k * 64 + c]      = (_Float16)w0;
    wq[k * 64 + 32 + c] = (_Float16)w1;
    float s = w0 * w0 + w1 * w1;
    #pragma unroll
    for (int m = 16; m >= 1; m >>= 1) s += __shfl_xor(s, m, 32);
    if (c == 0) norms[k] = s;
}

// ---------------------------------------------------------------------------
// Kernel 2: transpose inputs (B,C,H,W) f32 -> x_f16[row=(b,h,w)][c] row-major.
// grid = B*H blocks of 256 threads; each block handles one (b,h) slab.
__global__ void vq_transpose(const float* __restrict__ in,
                             _Float16* __restrict__ xq) {
    __shared__ _Float16 tile[64][33];          // pad to dodge bank conflicts
    int b = blockIdx.x >> 5;
    int h = blockIdx.x & 31;
    int tid = threadIdx.x;
    // read: coalesced along w within each channel row
    for (int j = tid; j < 2048; j += 256) {
        int c = j >> 5, w = j & 31;
        tile[c][w] = (_Float16)in[((b * 64 + c) * 32 + h) * 32 + w];
    }
    __syncthreads();
    // write: contiguous 64 halves (128B) per (b,h,w) row
    for (int j = tid; j < 2048; j += 256) {
        int w = j >> 6, c = j & 63;
        xq[(size_t)((b * 1024 + h * 32 + w)) * 64 + c] = tile[c][w];
    }
}

// ---------------------------------------------------------------------------
// Kernel 3: WMMA distance argmin + gather + loss.
// grid = 256 blocks of 256 threads (8 waves).
// Each wave owns TWO 16-row M-tiles (32 rows); block covers 256 rows of one b.
// B fragments are double-buffered (tile t+1 in flight during tile t's WMMAs),
// and the 4 WMMAs per tile use independent accumulators (no WMMA->WMMA RAW;
// the k-step combine is folded into the score FMA chain on the VALU, which
// co-executes with the XDL pipe).
__global__ void vq_main(const float* __restrict__ in,
                        const float* __restrict__ wgt,
                        const _Float16* __restrict__ xq,
                        const _Float16* __restrict__ wq,
                        const float* __restrict__ norms,
                        float* __restrict__ out_q,
                        float* __restrict__ out_idx,
                        float* __restrict__ out_loss) {
    __shared__ int   s_idx[256];
    __shared__ float s_red[256];

    const int tid  = threadIdx.x;
    const int wv   = tid >> 5;
    const int lane = tid & 31;
    const int m    = lane & 15;          // row-within-tile / column-within-tile
    const int hi   = lane >> 4;          // 0: lanes 0-15, 1: lanes 16-31

    // ---- load A fragments: 2 M-tiles x (16 rows x 64 c), two K=32 steps ----
    const int rowbase = blockIdx.x * 256 + wv * 32;
    const int coA = hi ? 16 : 0; // A layout: 16 consecutive K per lane (matches B)
    const _Float16* xp0 = xq + (size_t)(rowbase + m) * 64;        // M-tile 0
    const _Float16* xp1 = xp0 + 16 * 64;                          // M-tile 1
    V16 A0, A1, A2, A3;
    A0.h[0] = *(const v8h*)(xp0 + coA);
    A0.h[1] = *(const v8h*)(xp0 + coA + 8);
    A1.h[0] = *(const v8h*)(xp0 + 32 + coA);
    A1.h[1] = *(const v8h*)(xp0 + 40 + coA);
    A2.h[0] = *(const v8h*)(xp1 + coA);
    A2.h[1] = *(const v8h*)(xp1 + coA + 8);
    A3.h[0] = *(const v8h*)(xp1 + 32 + coA);
    A3.h[1] = *(const v8h*)(xp1 + 40 + coA);

    float minv[16];
    int   mini[16];
    #pragma unroll
    for (int r = 0; r < 16; ++r) { minv[r] = 3.4e38f; mini[r] = 0; }

    const int coB = hi ? 16 : 0;         // B layout: 16 consecutive K per lane (same as A)
    // ---- prologue: load B fragments + norm for tile 0 ----
    V16 B0, B1;
    {
        const _Float16* wp = wq + (size_t)m * 64 + coB;
        B0.h[0] = *(const v8h*)(wp);
        B0.h[1] = *(const v8h*)(wp + 8);
        B1.h[0] = *(const v8h*)(wp + 32);
        B1.h[1] = *(const v8h*)(wp + 40);
    }
    float nrm = norms[m];

    for (int t = 0; t < 64; ++t) {       // 64 column tiles of 16 codes
        // ---- prefetch tile t+1 (wraps to 0 on last iter; harmless) ----
        const int kn = (((t + 1) & 63) << 4) + m;
        const _Float16* wpn = wq + (size_t)kn * 64 + coB;
        V16 Bn0, Bn1;
        Bn0.h[0] = *(const v8h*)(wpn);
        Bn0.h[1] = *(const v8h*)(wpn + 8);
        Bn1.h[0] = *(const v8h*)(wpn + 32);
        Bn1.h[1] = *(const v8h*)(wpn + 40);
        const float nrmn = norms[kn];

        // ---- 4 independent WMMAs: 2 M-tiles x 2 K-steps ----
        v8f p0 = {}, p1 = {}, q0 = {}, q1 = {};
        p0 = __builtin_amdgcn_wmma_f32_16x16x32_f16(
            false, A0.v, false, B0.v, (short)0, p0, false, false);
        q0 = __builtin_amdgcn_wmma_f32_16x16x32_f16(
            false, A2.v, false, B0.v, (short)0, q0, false, false);
        p1 = __builtin_amdgcn_wmma_f32_16x16x32_f16(
            false, A1.v, false, B1.v, (short)0, p1, false, false);
        q1 = __builtin_amdgcn_wmma_f32_16x16x32_f16(
            false, A3.v, false, B1.v, (short)0, q1, false, false);

        const int k = (t << 4) + m;      // this lane's code column
        #pragma unroll
        for (int r = 0; r < 8; ++r) {
            // ||e||^2 - 2*(x.e_step0 + x.e_step1), folded as two FMAs
            float s0 = fmaf(-2.0f, p0[r], fmaf(-2.0f, p1[r], nrm));
            float s1 = fmaf(-2.0f, q0[r], fmaf(-2.0f, q1[r], nrm));
            if (s0 < minv[r])     { minv[r]     = s0; mini[r]     = k; }
            if (s1 < minv[8 + r]) { minv[8 + r] = s1; mini[8 + r] = k; }
        }

        B0 = Bn0; B1 = Bn1; nrm = nrmn;  // rotate double buffer
    }

    // ---- cross-lane min within each 16-lane half (columns of the D tile) ----
    #pragma unroll
    for (int r = 0; r < 16; ++r) {
        float mv = minv[r];
        int   mi = mini[r];
        #pragma unroll
        for (int mask = 8; mask >= 1; mask >>= 1) {
            float ov = __shfl_xor(mv, mask, 32);
            int   oi = __shfl_xor(mi, mask, 32);
            if (ov < mv || (ov == mv && oi < mi)) { mv = ov; mi = oi; }
        }
        minv[r] = mv; mini[r] = mi;
    }
    if (m == 0) {                         // lanes 0 and 16 publish their rows
        const int mb0 = wv * 32 + hi * 8;        // M-tile 0 rows
        const int mb1 = wv * 32 + 16 + hi * 8;   // M-tile 1 rows
        #pragma unroll
        for (int r = 0; r < 8; ++r) {
            s_idx[mb0 + r] = mini[r];
            s_idx[mb1 + r] = mini[8 + r];
        }
    }
    __syncthreads();

    // ---- gather quantized, write outputs, accumulate loss ----
    const int b   = blockIdx.x >> 2;
    const int hw0 = (blockIdx.x & 3) * 256;
    float partial = 0.0f;
    for (int j = tid; j < 64 * 256; j += 256) {
        const int c  = j >> 8;
        const int lr = j & 255;           // consecutive threads -> coalesced hw
        const int k  = s_idx[lr];
        const float wvv = wgt[k * 64 + c];
        const size_t off = (size_t)(b * 64 + c) * 1024 + hw0 + lr;
        const float xv = in[off];
        const float d  = wvv - xv;
        partial += d * d;
        out_q[off] = wvv;                 // quantized_st == quantized in value
        if (c == 0) out_idx[b * 1024 + hw0 + lr] = (float)k;
    }

    s_red[tid] = partial;
    __syncthreads();
    #pragma unroll
    for (int off = 128; off > 0; off >>= 1) {
        if (tid < off) s_red[tid] += s_red[tid + off];
        __syncthreads();
    }
    if (tid == 0) {
        // loss[b] = (1 + COMMITMENT_COST) * mean over C*H*W
        atomicAdd(&out_loss[b], s_red[0] * (1.25f / 65536.0f));
    }
}

// ---------------------------------------------------------------------------
extern "C" void kernel_launch(void* const* d_in, const int* in_sizes, int n_in,
                              void* d_out, int out_size, void* d_ws, size_t ws_size,
                              hipStream_t stream) {
    (void)in_sizes; (void)n_in; (void)out_size; (void)ws_size;
    const float* in  = (const float*)d_in[0];   // (64,64,32,32) f32
    const float* wgt = (const float*)d_in[1];   // (1024,64) f32

    float* out_q    = (float*)d_out;            // quantized_st, B*C*H*W
    float* out_idx  = out_q + OUT_Q;            // indices (as float), B*H*W
    float* out_loss = out_idx + OUT_I;          // loss, B

    // workspace: [0,128KB) f16 weight | [128KB,132KB) norms | [132KB,+8MB) f16 x
    _Float16* wq    = (_Float16*)d_ws;
    float*    norms = (float*)((char*)d_ws + 128 * 1024);
    _Float16* xq    = (_Float16*)((char*)d_ws + 132 * 1024);

    vq_prep<<<dim3(VQ_K), dim3(32), 0, stream>>>(wgt, wq, norms);
    vq_transpose<<<dim3(VQ_B * 32), dim3(256), 0, stream>>>(in, xq);
    hipMemsetAsync(out_loss, 0, VQ_B * sizeof(float), stream);
    vq_main<<<dim3(256), dim3(256), 0, stream>>>(in, wgt, xq, wq, norms,
                                                 out_q, out_idx, out_loss);
}
